// MPS_66726611911426
// MI455X (gfx1250) — matrix operand order repaired and checked by reading, not compiled, not run on hardware
//
#include <hip/hip_runtime.h>
#include <hip/hip_bf16.h>
#include <math.h>

typedef __attribute__((ext_vector_type(16))) _Float16 v16h;
typedef __attribute__((ext_vector_type(8)))  _Float16 v8h;
typedef __attribute__((ext_vector_type(8)))  float    v8f;
typedef __attribute__((ext_vector_type(4)))  float    v4f;

#define TT      254     // INPUT_SIZE - 2 scan steps
#define AL      128     // ALPHA
#define NCAT    4
#define NSAMP   1024
#define SPB     16      // samples per block (one 16-row WMMA tile)
#define USTRIDE 136     // padded LDS row stride in halfs (272B) -> conflict-free b128 reads

// ---------------------------------------------------------------------------
// Kernel 1: E = (f16) exp(matrices). 33 MB f16 -> resident in 192 MB L2.
// ---------------------------------------------------------------------------
__global__ void mps_exp_kernel(const float* __restrict__ M, _Float16* __restrict__ E, int n8) {
    int t = blockIdx.x * blockDim.x + threadIdx.x;
    if (t >= n8) return;
    const float* s = M + (size_t)t * 8;
    v4f a = *(const v4f*)(s);
    v4f b = *(const v4f*)(s + 4);
    v8h o;
    o[0] = (_Float16)__expf(a[0]); o[1] = (_Float16)__expf(a[1]);
    o[2] = (_Float16)__expf(a[2]); o[3] = (_Float16)__expf(a[3]);
    o[4] = (_Float16)__expf(b[0]); o[5] = (_Float16)__expf(b[1]);
    o[6] = (_Float16)__expf(b[2]); o[7] = (_Float16)__expf(b[3]);
    *(v8h*)(E + (size_t)t * 8) = o;
}

// ---------------------------------------------------------------------------
// Kernel 2: partition function log_z (sequential 254-step scan, 128 threads).
// Mz = log(sum_c exp(M_c)) handled in linear space: sum_c E_c, with running
// max-normalized state uz[] and scalar carry.
// ---------------------------------------------------------------------------
__global__ void mps_logz_kernel(const _Float16* __restrict__ E,
                                const float* __restrict__ row,
                                const float* __restrict__ column,
                                float* __restrict__ logz_out) {
    __shared__ float uz[AL];
    __shared__ float red[4];
    __shared__ float bval;
    const int i = threadIdx.x;          // 0..127
    const int lane = i & 31, w = i >> 5;
    const size_t catStride = (size_t)TT * AL * AL;

    // init from column: vz0[i] = log(sum_c exp(column[c,i]))
    float acc0 = 0.f;
    for (int c = 0; c < NCAT; ++c) acc0 += __expf(column[c * AL + i]);
    float s = __logf(acc0);
    float m = s;
    for (int d = 16; d; d >>= 1) m = fmaxf(m, __shfl_xor(m, d, 32));
    if (lane == 0) red[w] = m;
    __syncthreads();
    if (i == 0) bval = fmaxf(fmaxf(red[0], red[1]), fmaxf(red[2], red[3]));
    __syncthreads();
    float carry = bval;
    uz[i] = __expf(s - bval);
    __syncthreads();

    for (int k = TT - 1; k >= 0; --k) {
        float acc = 0.f;
        const _Float16* base = E + (size_t)k * AL * AL + (size_t)i * AL;
        for (int j = 0; j < AL; j += 8) {
            v8h e0 = *(const v8h*)(base + 0 * catStride + j);
            v8h e1 = *(const v8h*)(base + 1 * catStride + j);
            v8h e2 = *(const v8h*)(base + 2 * catStride + j);
            v8h e3 = *(const v8h*)(base + 3 * catStride + j);
#pragma unroll
            for (int q = 0; q < 8; ++q) {
                float ez = (float)e0[q] + (float)e1[q] + (float)e2[q] + (float)e3[q];
                acc += ez * uz[j + q];
            }
        }
        float mm = acc;
        for (int d = 16; d; d >>= 1) mm = fmaxf(mm, __shfl_xor(mm, d, 32));
        if (lane == 0) red[w] = mm;
        __syncthreads();
        if (i == 0) bval = fmaxf(fmaxf(red[0], red[1]), fmaxf(red[2], red[3]));
        __syncthreads();
        float mx = bval;
        uz[i] = acc / mx;
        if (i == 0) carry += __logf(mx);
        __syncthreads();
    }
    // log_z = carry + log(sum_i (sum_c exp(row[c,i])) * uz[i])
    float e = 0.f;
    for (int c = 0; c < NCAT; ++c) e += __expf(row[c * AL + i]);
    float term = e * uz[i];
    for (int d = 16; d; d >>= 1) term += __shfl_xor(term, d, 32);
    if (lane == 0) red[w] = term;
    __syncthreads();
    if (i == 0) *logz_out = carry + __logf(red[0] + red[1] + red[2] + red[3]);
}

// ---------------------------------------------------------------------------
// Kernel 3: main scan. 64 blocks x 16 samples, 8 waves/block.
// Wave w -> category c = w>>1, column half h = w&1 (64 alphas).
// Per step: D(16x16,f32) += A(u tile 16x32 f16, LDS) x B(E^T 32x16 f16, L2),
// 16 v_wmma per wave per step; then per-sample max-normalize + category select.
// ---------------------------------------------------------------------------
__global__ void __launch_bounds__(256)
mps_main_kernel(const int* __restrict__ x, const _Float16* __restrict__ E,
                const float* __restrict__ row, const float* __restrict__ col,
                const float* __restrict__ logz, float* __restrict__ out) {
    __shared__ _Float16 U[SPB * USTRIDE];        // u[n,j] = exp(v - carry), f16
    __shared__ float smax[NCAT][2][SPB];         // per (cat, half) row maxima
    __shared__ float carry[SPB];
    __shared__ unsigned char xs[SPB][256];       // per-sample categories

    const int tid  = threadIdx.x;
    const int lane = tid & 31;
    const int w    = tid >> 5;        // wave 0..7
    const int c    = w >> 1;          // category computed by this wave
    const int h    = w & 1;           // column half (0: i<64, 1: i>=64)
    const int s0   = blockIdx.x * SPB;

    // stage the 16 samples' category rows
    for (int idx = tid; idx < SPB * 256; idx += blockDim.x) {
        int sm = idx >> 8, cc = idx & 255;
        xs[sm][cc] = (unsigned char)x[(s0 + sm) * 256 + cc];
    }
    __syncthreads();

    // init state: v0 = column[x[:,255]]; carry = max; u = exp(v0 - carry)
    if (tid < SPB) {
        const float* vc = col + (int)xs[tid][255] * AL;
        float mx = -3.0e38f;
        for (int j = 0; j < AL; ++j) mx = fmaxf(mx, vc[j]);
        carry[tid] = mx;
        for (int j = 0; j < AL; ++j) U[tid * USTRIDE + j] = (_Float16)__expf(vc[j] - mx);
    }
    __syncthreads();

    const int colInTile = lane & 15;
    const int half32    = lane >> 4;      // lane group (ISA §7.12.2 layouts)
    const int rowA      = colInTile;      // A-matrix row = sample
    const int aoff      = half32 * 16;    // A: K offset (halfs)
    const int boff      = half32 * 16;    // B: K offset (halfs)
    const v8f vzero = {0.f, 0.f, 0.f, 0.f, 0.f, 0.f, 0.f, 0.f};

    for (int k = TT - 1; k >= 0; --k) {
        const _Float16* Ebase = E + ((size_t)c * TT + k) * (AL * AL);
        v8f acc[4] = {vzero, vzero, vzero, vzero};
#pragma unroll
        for (int kt = 0; kt < 4; ++kt) {
            const int jb = kt * 32;
            // A fragment (shared across this wave's 4 tiles): u[sample, jb..jb+31]
            v8h alo = *(const v8h*)&U[rowA * USTRIDE + jb + aoff];
            v8h ahi = *(const v8h*)&U[rowA * USTRIDE + jb + aoff + 8];
            v16h a = __builtin_shufflevector(alo, ahi,
                    0, 1, 2, 3, 4, 5, 6, 7, 8, 9, 10, 11, 12, 13, 14, 15);
#pragma unroll
            for (int tile = 0; tile < 4; ++tile) {
                const int i = h * 64 + tile * 16 + colInTile;   // output alpha
                // B[j,i] = E[i,j]: 16 contiguous halfs of E row i, 32B aligned
                v16h b = *(const v16h*)(Ebase + (size_t)i * AL + jb + boff);
                acc[tile] = __builtin_amdgcn_wmma_f32_16x16x32_f16(
                    false, a, false, b, (short)0, acc[tile], false, false);
            }
        }
        // hint next step's matrix rows toward the caches (k-1)
        if (k > 0)
            __builtin_prefetch(Ebase - (AL * AL) + (size_t)(h * 64 + colInTile) * AL, 0, 1);

        // per-sample max over this wave's 64 columns (cols live in lanes)
        if (colInTile == 0) { /* placeholder to keep EXEC uniform before shuffles */ }
#pragma unroll
        for (int r = 0; r < 8; ++r) {
            float v = fmaxf(fmaxf(acc[0][r], acc[1][r]), fmaxf(acc[2][r], acc[3][r]));
            v = fmaxf(v, __shfl_xor(v, 1, 32));
            v = fmaxf(v, __shfl_xor(v, 2, 32));
            v = fmaxf(v, __shfl_xor(v, 4, 32));
            v = fmaxf(v, __shfl_xor(v, 8, 32));
            if (colInTile == 0) smax[c][h][r + half32 * 8] = v;
        }
        __syncthreads();

        // category select + max-normalize + write next u tile
        const int xcol = k + 1;
#pragma unroll
        for (int r = 0; r < 8; ++r) {
            const int samp = r + half32 * 8;
            if ((int)xs[samp][xcol] == c) {
                float mx  = fmaxf(smax[c][0][samp], smax[c][1][samp]);
                float rin = 1.0f / mx;
#pragma unroll
                for (int tile = 0; tile < 4; ++tile)
                    U[samp * USTRIDE + h * 64 + tile * 16 + colInTile] =
                        (_Float16)(acc[tile][r] * rin);
            }
        }
        if (w == 0 && lane < SPB) {
            int cn = xs[lane][xcol];
            carry[lane] += __logf(fmaxf(smax[cn][0][lane], smax[cn][1][lane]));
        }
        __syncthreads();
    }

    // finalize: lp = carry + log(sum_j exp(row[x0,j]) * u[j]) - log_z
    if (tid < SPB) {
        const float* rp = row + (int)xs[tid][0] * AL;
        float ssum = 0.f;
        for (int j = 0; j < AL; ++j)
            ssum += __expf(rp[j]) * (float)U[tid * USTRIDE + j];
        out[s0 + tid] = carry[tid] + __logf(ssum) - logz[0];
    }
}

// ---------------------------------------------------------------------------
extern "C" void kernel_launch(void* const* d_in, const int* in_sizes, int n_in,
                              void* d_out, int out_size, void* d_ws, size_t ws_size,
                              hipStream_t stream) {
    const int*   x   = (const int*)d_in[0];     // (N, 256) int32
    const float* M   = (const float*)d_in[1];   // (4, 254, 128, 128) f32
    const float* row = (const float*)d_in[2];   // (4, 1, 128) f32
    const float* col = (const float*)d_in[3];   // (4, 128, 1) f32
    float* out = (float*)d_out;                 // (N,) f32

    _Float16* E = (_Float16*)d_ws;              // 33.3 MB: exp(M) in f16
    const size_t nE = (size_t)NCAT * TT * AL * AL;
    float* logz = (float*)((char*)d_ws + nE * sizeof(_Float16));

    const int n8 = (int)(nE / 8);
    mps_exp_kernel<<<(n8 + 255) / 256, 256, 0, stream>>>(M, E, n8);
    mps_logz_kernel<<<1, 128, 0, stream>>>(E, row, col, logz);
    mps_main_kernel<<<NSAMP / SPB, 256, 0, stream>>>(x, E, row, col, logz, out);
}
